// RespectiveLayer_29669634080804
// MI455X (gfx1250) — matrix operand order repaired and checked by reading, not compiled, not run on hardware
//
#include <hip/hip_runtime.h>
#include <hip/hip_fp16.h>
#include <hip/hip_bf16.h>

typedef __attribute__((ext_vector_type(16))) _Float16 v16h;
typedef __attribute__((ext_vector_type(8)))  _Float16 v8h;
typedef __attribute__((ext_vector_type(8)))  float    v8f;

#define DME 768
#define QKVD 2304

// ---------------------------------------------------------------------------
// Generic f32 -> f16 conversion
// ---------------------------------------------------------------------------
__global__ void k_f32_to_f16(const float* __restrict__ in, _Float16* __restrict__ out,
                             long long n) {
  long long t = (long long)blockIdx.x * blockDim.x + threadIdx.x;
  if (t < n) out[t] = (_Float16)in[t];
}

// ---------------------------------------------------------------------------
// LayerNorm over rows of 768, f32 in, f16 out (feeds WMMA GEMM)
// ---------------------------------------------------------------------------
__global__ void __launch_bounds__(256)
k_layernorm_f16(const float* __restrict__ x, const float* __restrict__ g,
                const float* __restrict__ b, _Float16* __restrict__ out, int rows) {
  __shared__ float s1[256], s2[256];
  const int row = blockIdx.x;
  if (row >= rows) return;
  const float* xr = x + (size_t)row * DME;
  float sum = 0.f, sq = 0.f;
  for (int i = threadIdx.x; i < DME; i += 256) { float v = xr[i]; sum += v; sq += v * v; }
  s1[threadIdx.x] = sum; s2[threadIdx.x] = sq;
  __syncthreads();
  for (int off = 128; off > 0; off >>= 1) {
    if ((int)threadIdx.x < off) {
      s1[threadIdx.x] += s1[threadIdx.x + off];
      s2[threadIdx.x] += s2[threadIdx.x + off];
    }
    __syncthreads();
  }
  const float mu  = s1[0] * (1.0f / DME);
  const float var = s2[0] * (1.0f / DME) - mu * mu;
  const float r   = rsqrtf(var + 1e-5f);
  for (int i = threadIdx.x; i < DME; i += 256)
    out[(size_t)row * DME + i] = (_Float16)((xr[i] - mu) * r * g[i] + b[i]);
}

// ---------------------------------------------------------------------------
// WMMA GEMM: C[M,N] = A[M,K](f16) @ W[N,K](f16)^T (+bias +resid, opt leaky)
// One wave computes a 16x64 strip; 4 accumulators reuse the A fragment.
// ---------------------------------------------------------------------------
__global__ void __launch_bounds__(256)
k_gemm_wmma(const _Float16* __restrict__ A, const _Float16* __restrict__ W,
            const float* __restrict__ bias, const float* __restrict__ resid,
            float* __restrict__ Cf, _Float16* __restrict__ Ch,
            int M, int Nn, int Kk, int act) {
  const int numNt = Nn >> 6;
  const int wave  = (int)(((long long)blockIdx.x * blockDim.x + threadIdx.x) >> 5);
  const int lane  = threadIdx.x & 31;
  const int total = (M >> 4) * numNt;
  if (wave >= total) return;                 // wave-uniform: EXEC stays all-ones
  const int mt   = wave / numNt;
  const int nt   = wave - mt * numNt;
  const int half = lane >> 4;                // 0/1 lane-group (ISA A/B K split)
  const int l16  = lane & 15;

  const _Float16* __restrict__ Arow = A + (size_t)(mt * 16 + l16) * Kk + half * 16;
  const _Float16* __restrict__ B0   = W + (size_t)(nt * 64 + l16) * Kk + half * 16;
  const size_t bstride = (size_t)16 * Kk;

  v8f acc0 = {}, acc1 = {}, acc2 = {}, acc3 = {};
  for (int k0 = 0; k0 < Kk; k0 += 32) {
    v16h a;
    {
      v8h alo = *(const v8h*)(Arow + k0);        // K = kb..kb+7
      v8h ahi = *(const v8h*)(Arow + k0 + 8);    // K = kb+8..kb+15
#pragma unroll
      for (int i = 0; i < 8; ++i) { a[i] = alo[i]; a[i + 8] = ahi[i]; }
    }
    v16h b0 = *(const v16h*)(B0 + k0);
    v16h b1 = *(const v16h*)(B0 + bstride + k0);
    v16h b2 = *(const v16h*)(B0 + 2 * bstride + k0);
    v16h b3 = *(const v16h*)(B0 + 3 * bstride + k0);
    acc0 = __builtin_amdgcn_wmma_f32_16x16x32_f16(false, a, false, b0, (short)0, acc0, false, false);
    acc1 = __builtin_amdgcn_wmma_f32_16x16x32_f16(false, a, false, b1, (short)0, acc1, false, false);
    acc2 = __builtin_amdgcn_wmma_f32_16x16x32_f16(false, a, false, b2, (short)0, acc2, false, false);
    acc3 = __builtin_amdgcn_wmma_f32_16x16x32_f16(false, a, false, b3, (short)0, acc3, false, false);
  }

  const int row0 = mt * 16 + half * 8;       // C/D layout: VGPR r -> row r + 8*half
#pragma unroll
  for (int j = 0; j < 4; ++j) {
    v8f acc = (j == 0) ? acc0 : (j == 1) ? acc1 : (j == 2) ? acc2 : acc3;
    const int col = nt * 64 + j * 16 + l16;
    const float bv = bias ? bias[col] : 0.0f;
#pragma unroll
    for (int r = 0; r < 8; ++r) {
      const int row = row0 + r;
      float v = acc[r] + bv;
      if (resid) v += resid[(size_t)row * Nn + col];
      if (act)   v = v > 0.0f ? v : 0.01f * v;   // LeakyReLU(0.01)
      if (Cf) Cf[(size_t)row * Nn + col] = v;
      if (Ch) Ch[(size_t)row * Nn + col] = (_Float16)v;
    }
  }
}

// ---------------------------------------------------------------------------
// LModel attention pieces (B=8,S=126,H=4,hd=192) — sub-GFLOP, VALU kernels
// ---------------------------------------------------------------------------
__global__ void k_scores_l(const float* __restrict__ qkv, float* __restrict__ sc) {
  int t = blockIdx.x * blockDim.x + threadIdx.x;
  if (t >= 8 * 4 * 126 * 126) return;
  int ki = t % 126; int r = t / 126;
  int qi = r % 126; r /= 126;
  int h = r % 4; int bb = r / 4;
  const float* q = qkv + (size_t)(bb * 126 + qi) * QKVD + h * 192;
  const float* k = qkv + (size_t)(bb * 126 + ki) * QKVD + 768 + h * 192;
  float s = 0.f;
  for (int d = 0; d < 192; ++d) s += q[d] * k[d];
  sc[t] = s * 0.07216878365f;                 // 1/sqrt(192)
}

__global__ void k_softmax_rows(float* __restrict__ x, int rows, int L) {
  int wv = (int)(((long long)blockIdx.x * blockDim.x + threadIdx.x) >> 5);
  int lane = threadIdx.x & 31;
  if (wv >= rows) return;
  float* row = x + (size_t)wv * L;
  float mx = -3.4e38f;
  for (int i = lane; i < L; i += 32) mx = fmaxf(mx, row[i]);
  for (int o = 16; o > 0; o >>= 1) mx = fmaxf(mx, __shfl_xor(mx, o, 32));
  float sum = 0.f;
  for (int i = lane; i < L; i += 32) { float e = __expf(row[i] - mx); row[i] = e; sum += e; }
  for (int o = 16; o > 0; o >>= 1) sum += __shfl_xor(sum, o, 32);
  float inv = 1.0f / sum;
  for (int i = lane; i < L; i += 32) row[i] *= inv;
}

__global__ void k_ctx_l(const float* __restrict__ qkv, const float* __restrict__ attn,
                        _Float16* __restrict__ out) {
  int t = blockIdx.x * blockDim.x + threadIdx.x;
  if (t >= 1008 * DME) return;
  int d = t % DME; int row = t / DME;
  int bb = row / 126, qi = row % 126;
  int h = d / 192;
  const float* a = attn + ((size_t)(bb * 4 + h) * 126 + qi) * 126;
  const float* v = qkv + (size_t)bb * 126 * QKVD + 1536 + d;
  float s = 0.f;
  for (int kk = 0; kk < 126; ++kk) s += a[kk] * v[(size_t)kk * QKVD];
  out[t] = (_Float16)s;
}

// attn.mean(heads) then 21x21 adaptive avg-pool -> [B,6,6]
__global__ void k_pool(const float* __restrict__ attn, float* __restrict__ out) {
  int t = blockIdx.x * blockDim.x + threadIdx.x;
  if (t >= 8 * 36) return;
  int j = t % 6, i = (t / 6) % 6, bb = t / 36;
  float s = 0.f;
  for (int h = 0; h < 4; ++h)
    for (int q = i * 21; q < i * 21 + 21; ++q)
      for (int k = j * 21; k < j * 21 + 21; ++k)
        s += attn[((size_t)(bb * 4 + h) * 126 + q) * 126 + k];
  out[t] = s * (1.0f / (4.0f * 441.0f));
}

// ---------------------------------------------------------------------------
// GCN: degree, symmetric norm, init (self-loop + bias), edge scatter-add
// ---------------------------------------------------------------------------
__global__ void k_zero_f32(float* __restrict__ p, long long n) {
  long long t = (long long)blockIdx.x * blockDim.x + threadIdx.x;
  if (t < n) p[t] = 0.f;
}

__global__ void k_deg(const int* __restrict__ dst, float* __restrict__ deg, int E) {
  int t = blockIdx.x * blockDim.x + threadIdx.x;
  if (t >= E) return;
  __hip_atomic_fetch_add(&deg[dst[t]], 1.0f, __ATOMIC_RELAXED, __HIP_MEMORY_SCOPE_AGENT);
}

__global__ void k_dinv(float* __restrict__ deg, int n) {
  int t = blockIdx.x * blockDim.x + threadIdx.x;
  if (t < n) deg[t] = rsqrtf(deg[t] + 1.0f);   // +1 self-loop; deg>=1 guaranteed
}

__global__ void k_feats_init(const float* __restrict__ h, const float* __restrict__ dinv,
                             const float* __restrict__ bg, float* __restrict__ feats,
                             long long n) {
  long long t = (long long)blockIdx.x * blockDim.x + threadIdx.x;
  if (t >= n) return;
  int i = (int)(t / DME), d = (int)(t % DME);
  float w = dinv[i];
  feats[t] = bg[d] + h[t] * w * w;             // self-loop term + bias
}

__global__ void k_edge_agg(const int* __restrict__ src, const int* __restrict__ dst,
                           const float* __restrict__ h, const float* __restrict__ dinv,
                           float* __restrict__ feats, long long total) {
  long long t = (long long)blockIdx.x * blockDim.x + threadIdx.x;
  if (t >= total) return;
  int e = (int)(t / 192);
  int q = (int)(t - (long long)e * 192) * 4;   // 4 floats per thread, coalesced
  int s = src[e], d = dst[e];
  float w = dinv[s] * dinv[d];
  const float* hp = h + (size_t)s * DME + q;
  float* op = feats + (size_t)d * DME + q;
#pragma unroll
  for (int c = 0; c < 4; ++c)
    __hip_atomic_fetch_add(op + c, hp[c] * w, __ATOMIC_RELAXED, __HIP_MEMORY_SCOPE_AGENT);
}

// ---------------------------------------------------------------------------
// MultiAttn: gather neighborhoods, center-row attention, write back centers
// ---------------------------------------------------------------------------
__global__ void k_gather_f16(const float* __restrict__ feats, const int* __restrict__ idx,
                             _Float16* __restrict__ out, long long n) {
  long long t = (long long)blockIdx.x * blockDim.x + threadIdx.x;
  if (t >= n) return;
  int d = (int)(t % DME); long long r = t / DME;    // r = u*32 + j
  int node = idx[r];
  out[t] = (_Float16)feats[(size_t)node * DME + d];
}

__global__ void k_scores_m(const float* __restrict__ qkv, float* __restrict__ sc) {
  int t = blockIdx.x * blockDim.x + threadIdx.x;
  if (t >= 1024 * 4 * 32) return;
  int j = t % 32; int r = t / 32; int h = r % 4; int u = r / 4;
  const float* q = qkv + (size_t)(u * 32) * QKVD + h * 192;            // center row only
  const float* k = qkv + (size_t)(u * 32 + j) * QKVD + 768 + h * 192;
  float s = 0.f;
  for (int d = 0; d < 192; ++d) s += q[d] * k[d];
  sc[t] = s * 0.07216878365f;
}

__global__ void k_ctx_m(const float* __restrict__ qkv, const float* __restrict__ attn,
                        _Float16* __restrict__ out) {
  int t = blockIdx.x * blockDim.x + threadIdx.x;
  if (t >= 1024 * DME) return;
  int d = t % DME; int u = t / DME; int h = d / 192;
  const float* a = attn + (size_t)(u * 4 + h) * 32;
  const float* v = qkv + (size_t)(u * 32) * QKVD + 1536 + d;
  float s = 0.f;
  for (int j = 0; j < 32; ++j) s += a[j] * v[(size_t)j * QKVD];
  out[t] = (_Float16)s;
}

__global__ void k_scatter_center(const float* __restrict__ cm, const int* __restrict__ idx,
                                 float* __restrict__ feats) {
  int t = blockIdx.x * blockDim.x + threadIdx.x;
  if (t >= 1024 * DME) return;
  int d = t % DME; int u = t / DME;
  int node = idx[u * 32];
  feats[(size_t)node * DME + d] = cm[t];
}

// ---------------------------------------------------------------------------
static inline int nblk(long long n, int bs) { return (int)((n + bs - 1) / bs); }

extern "C" void kernel_launch(void* const* d_in, const int* in_sizes, int n_in,
                              void* d_out, int out_size, void* d_ws, size_t ws_size,
                              hipStream_t stream) {
  (void)in_sizes; (void)n_in; (void)out_size; (void)ws_size;
  const int D = 768, QKV = 2304, Bb = 8, S = 126, BS = Bb * S, H = 4;
  const int N = 100000, E = 1600000, U = 1024, Kn = 32, UK = U * Kn;

  // ---- inputs (setup_inputs order) ----
  const float* user_text = (const float*)d_in[0];
  const float* all_feat  = (const float*)d_in[1];
  const int*   nbr_idx   = (const int*)d_in[2];
  const int*   edge      = (const int*)d_in[3];      // [2,E]: src then dst
  const float* Wqkv_l = (const float*)d_in[4];  const float* bqkv_l = (const float*)d_in[5];
  const float* Wo_l   = (const float*)d_in[6];  const float* bo_l   = (const float*)d_in[7];
  const float* ln1_g  = (const float*)d_in[8];  const float* ln1_b  = (const float*)d_in[9];
  const float* ln2_g  = (const float*)d_in[10]; const float* ln2_b  = (const float*)d_in[11];
  const float* W1 = (const float*)d_in[12]; const float* b1 = (const float*)d_in[13];
  const float* W2 = (const float*)d_in[14]; const float* b2 = (const float*)d_in[15];
  const float* Wg = (const float*)d_in[16]; const float* bg = (const float*)d_in[17];
  const float* Wqkv_m = (const float*)d_in[18]; const float* bqkv_m = (const float*)d_in[19];
  const float* Wo_m   = (const float*)d_in[20]; const float* bo_m   = (const float*)d_in[21];

  // ---- outputs: text[BS,D] ++ feats[N,D] ++ attn_graph[B,6,6] ----
  float* text_out  = (float*)d_out;
  float* feats_out = text_out + (size_t)BS * D;
  float* graph_out = feats_out + (size_t)N * D;

  // ---- workspace layout ----
  char* ws = (char*)d_ws;
  size_t off = 0;
  auto alloc = [&](size_t bytes) { size_t o = off; off += (bytes + 255) & ~(size_t)255; return o; };
  _Float16* wqkv_l_h = (_Float16*)(ws + alloc((size_t)QKV * D * 2));
  _Float16* wo_l_h   = (_Float16*)(ws + alloc((size_t)D * D * 2));
  _Float16* w1_h     = (_Float16*)(ws + alloc((size_t)D * D * 2));
  _Float16* w2_h     = (_Float16*)(ws + alloc((size_t)D * D * 2));
  _Float16* wg_h     = (_Float16*)(ws + alloc((size_t)D * D * 2));
  _Float16* wqkv_m_h = (_Float16*)(ws + alloc((size_t)QKV * D * 2));
  _Float16* wo_m_h   = (_Float16*)(ws + alloc((size_t)D * D * 2));
  _Float16* ln1_h    = (_Float16*)(ws + alloc((size_t)BS * D * 2));
  _Float16* ln2_h    = (_Float16*)(ws + alloc((size_t)BS * D * 2));
  _Float16* h1_h     = (_Float16*)(ws + alloc((size_t)BS * D * 2));
  _Float16* ctx_h    = (_Float16*)(ws + alloc((size_t)BS * D * 2));
  float* qkv_l    = (float*)(ws + alloc((size_t)BS * QKV * 4));
  float* scores_l = (float*)(ws + alloc((size_t)Bb * H * S * S * 4));
  float* text1    = (float*)(ws + alloc((size_t)BS * D * 4));
  float* deg      = (float*)(ws + alloc((size_t)N * 4));
  float* scores_m = (float*)(ws + alloc((size_t)U * H * Kn * 4));
  _Float16* ctx_m_h = (_Float16*)(ws + alloc((size_t)U * D * 2));
  float* cm       = (float*)(ws + alloc((size_t)U * D * 4));
  // big1: X_f16 (dead after GCN GEMM) union qkv_m
  size_t big1 = alloc((size_t)UK * QKV * 4);
  _Float16* X_h   = (_Float16*)(ws + big1);
  float*    qkv_m = (float*)(ws + big1);
  // big2: h_gcn (dead after edge agg) union nb_f16
  size_t big2 = alloc((size_t)N * D * 4);
  float*    h_gcn = (float*)(ws + big2);
  _Float16* nb_h  = (_Float16*)(ws + big2);

  auto conv = [&](const float* src, _Float16* dst, long long n) {
    k_f32_to_f16<<<nblk(n, 256), 256, 0, stream>>>(src, dst, n);
  };
  auto gemm = [&](const _Float16* A, const _Float16* Wt, const float* bias,
                  const float* resid, float* Cf, _Float16* Ch,
                  int M, int Nc, int Kc, int act) {
    long long waves = (long long)(M / 16) * (Nc / 64);
    k_gemm_wmma<<<nblk(waves * 32, 256), 256, 0, stream>>>(A, Wt, bias, resid, Cf, Ch,
                                                           M, Nc, Kc, act);
  };

  // ---- weights to f16 ----
  conv(Wqkv_l, wqkv_l_h, (long long)QKV * D);
  conv(Wo_l,   wo_l_h,   (long long)D * D);
  conv(W1,     w1_h,     (long long)D * D);
  conv(W2,     w2_h,     (long long)D * D);
  conv(Wg,     wg_h,     (long long)D * D);
  conv(Wqkv_m, wqkv_m_h, (long long)QKV * D);
  conv(Wo_m,   wo_m_h,   (long long)D * D);

  // ---- LModel: pre-norm transformer block ----
  k_layernorm_f16<<<BS, 256, 0, stream>>>(user_text, ln1_g, ln1_b, ln1_h, BS);
  gemm(ln1_h, wqkv_l_h, bqkv_l, nullptr, qkv_l, nullptr, BS, QKV, D, 0);
  k_scores_l<<<nblk((long long)Bb * H * S * S, 256), 256, 0, stream>>>(qkv_l, scores_l);
  k_softmax_rows<<<nblk((long long)Bb * H * S * 32, 256), 256, 0, stream>>>(scores_l, Bb * H * S, S);
  k_pool<<<nblk(8 * 36, 256), 256, 0, stream>>>(scores_l, graph_out);
  k_ctx_l<<<nblk((long long)BS * D, 256), 256, 0, stream>>>(qkv_l, scores_l, ctx_h);
  gemm(ctx_h, wo_l_h, bo_l, user_text, text1, nullptr, BS, D, D, 0);   // text1 = x + SA
  k_layernorm_f16<<<BS, 256, 0, stream>>>(text1, ln2_g, ln2_b, ln2_h, BS);
  gemm(ln2_h, w1_h, b1, nullptr, nullptr, h1_h, BS, D, D, 1);          // LeakyReLU, f16 out
  gemm(h1_h, w2_h, b2, text1, text_out, nullptr, BS, D, D, 0);         // final text

  // ---- GCN over user graph ----
  conv(all_feat, X_h, (long long)N * D);
  gemm(X_h, wg_h, nullptr, nullptr, h_gcn, nullptr, N, D, D, 0);       // h = X @ Wg^T
  k_zero_f32<<<nblk(N, 256), 256, 0, stream>>>(deg, N);
  k_deg<<<nblk(E, 256), 256, 0, stream>>>(edge + E, deg, E);           // indegree over dst
  k_dinv<<<nblk(N, 256), 256, 0, stream>>>(deg, N);                    // deg := rsqrt(deg+1)
  k_feats_init<<<nblk((long long)N * D, 256), 256, 0, stream>>>(h_gcn, deg, bg, feats_out,
                                                                (long long)N * D);
  k_edge_agg<<<nblk((long long)E * 192, 256), 256, 0, stream>>>(edge, edge + E, h_gcn, deg,
                                                                feats_out, (long long)E * 192);

  // ---- MultiAttn: center-row attention per neighborhood ----
  k_gather_f16<<<nblk((long long)UK * D, 256), 256, 0, stream>>>(feats_out, nbr_idx, nb_h,
                                                                 (long long)UK * D);
  gemm(nb_h, wqkv_m_h, bqkv_m, nullptr, qkv_m, nullptr, UK, QKV, D, 0);
  k_scores_m<<<nblk((long long)U * H * Kn, 256), 256, 0, stream>>>(qkv_m, scores_m);
  k_softmax_rows<<<nblk((long long)U * H * 32, 256), 256, 0, stream>>>(scores_m, U * H, Kn);
  k_ctx_m<<<nblk((long long)U * D, 256), 256, 0, stream>>>(qkv_m, scores_m, ctx_m_h);
  gemm(ctx_m_h, wo_m_h, bo_m, nullptr, cm, nullptr, U, D, D, 0);
  k_scatter_center<<<nblk((long long)U * D, 256), 256, 0, stream>>>(cm, nbr_idx, feats_out);
}
